// Seq2SeqAE_47742856462903
// MI455X (gfx1250) — matrix operand order repaired and checked by reading, not compiled, not run on hardware
//
#include <hip/hip_runtime.h>
#include <hip/hip_bf16.h>

// Seq2Seq GRU autoencoder for MI455X (gfx1250, wave32, WMMA).
// V=32000 E=256 H=512 L=128 B=16 T=128.
// All matrix work uses v_wmma_f32_16x16x32_bf16 (f32 accumulate).

#define V_  32000
#define E_  256
#define H_  512
#define L_  128
#define B_  16
#define T_  128
#define G3H 1536            // 3*H
#define NT3H 96             // G3H/16
#define NTV  2000           // V/16

typedef __bf16 bfx16 __attribute__((ext_vector_type(16)));
typedef float  v8f   __attribute__((ext_vector_type(8)));
typedef float  fx4   __attribute__((ext_vector_type(4)));

__device__ __forceinline__ v8f wmma_bf16(bfx16 a, bfx16 b, v8f c) {
  return __builtin_amdgcn_wmma_f32_16x16x32_bf16(false, a, false, b,
                                                 (short)0, c, false, false);
}
__device__ __forceinline__ float sig_(float x) { return 1.0f / (1.0f + __expf(-x)); }

// ---------------------------------------------------------------------------
// Pack a row-major f32 weight W[N][K] into bf16 B-operand tiles:
// layout [nt][ks][lane][16]; lane = n%16 + 16*half, element j -> K = ks*32 + 8*half + (j&7) + 16*(j>>3).
// ---------------------------------------------------------------------------
__global__ void k_pack_b(const float* __restrict__ W, __bf16* __restrict__ out,
                         int N, int K) {
  int Ksteps = K >> 5;
  long long total = (long long)N * K;
  long long o = (long long)blockIdx.x * blockDim.x + threadIdx.x;
  if (o >= total) return;
  int j    = (int)(o & 15);
  int lane = (int)((o >> 4) & 31);
  long long rest = o >> 9;
  int ks = (int)(rest % Ksteps);
  int nt = (int)(rest / Ksteps);
  int n  = nt * 16 + (lane & 15);
  int hf = lane >> 4;
  int k  = ks * 32 + hf * 8 + (j & 7) + ((j & 8) << 1);
  out[o] = (__bf16)W[(long long)n * K + k];
}

// ---------------------------------------------------------------------------
// gx = emb[x] @ W_ih^T + b_ih  for all timesteps. One wave per 16x16 tile.
// A operand gathered from embedding rows directly in the 16-bit A layout:
// lane (0..31): row m = lane&15, a[0..7] -> K = ks*32 + 8*half + j,
//               a[8..15] -> K = ks*32 + 16 + 8*half + (j-8).
// ---------------------------------------------------------------------------
__global__ __launch_bounds__(32)
void k_gx(const int* __restrict__ x,          // [B,T]
          const float* __restrict__ emb,      // [V,E]
          const __bf16* __restrict__ Wp,      // packed [96][8][32][16]
          const float* __restrict__ bias,     // [3H]
          float* __restrict__ gx) {           // [steps,B,3H]
  int nt = blockIdx.x;            // 0..95
  int t  = blockIdx.y;            // timestep
  int lane = threadIdx.x;
  int m  = lane & 15;             // batch row
  int hf = lane >> 4;
  int tok = x[m * T_ + t];
  const float* row = emb + (long long)tok * E_;
  v8f acc = {};
#pragma unroll
  for (int ks = 0; ks < E_ / 32; ++ks) {
    int b1 = ks * 32 + 8 * hf;
    fx4 a0 = *(const fx4*)(row + b1);
    fx4 a1 = *(const fx4*)(row + b1 + 4);
    fx4 a2 = *(const fx4*)(row + b1 + 16);
    fx4 a3 = *(const fx4*)(row + b1 + 20);
    bfx16 a;
    a[0]=(__bf16)a0.x; a[1]=(__bf16)a0.y; a[2]=(__bf16)a0.z; a[3]=(__bf16)a0.w;
    a[4]=(__bf16)a1.x; a[5]=(__bf16)a1.y; a[6]=(__bf16)a1.z; a[7]=(__bf16)a1.w;
    a[8]=(__bf16)a2.x; a[9]=(__bf16)a2.y; a[10]=(__bf16)a2.z; a[11]=(__bf16)a2.w;
    a[12]=(__bf16)a3.x; a[13]=(__bf16)a3.y; a[14]=(__bf16)a3.z; a[15]=(__bf16)a3.w;
    bfx16 b = *(const bfx16*)(Wp + ((long long)(nt * 8 + ks) * 32 + lane) * 16);
    acc = wmma_bf16(a, b, acc);
  }
  int n = nt * 16 + (lane & 15);
  float bb = bias[n];
#pragma unroll
  for (int r = 0; r < 8; ++r) {
    int mm = r + 8 * hf;          // batch
    gx[((long long)t * B_ + mm) * G3H + n] = acc[r] + bb;
  }
}

// ---------------------------------------------------------------------------
// Persistent GRU scan: one workgroup of 32 waves. Wave cb owns h columns
// [cb*16, cb*16+16) as an f32 C-layout register tile, and computes the r/z/n
// gh tiles (nt = cb, cb+32, cb+64) for those columns -> gate math is wave-local.
// h is re-packed to the bf16 A layout in LDS (16 KB) each step.
// ---------------------------------------------------------------------------
__global__ __launch_bounds__(1024)
void k_gru_scan(const float* __restrict__ gx,     // [steps,B,3H]
                const __bf16* __restrict__ Whh,   // packed [96][16][32][16]
                const float* __restrict__ bhh,    // [3H]
                const float* __restrict__ h0,     // [B,H] or nullptr (zeros)
                int steps,
                __bf16* __restrict__ hsA,         // [steps][8192] A-packed or nullptr
                float* __restrict__ hOut) {       // [B,H] or nullptr
  __shared__ __bf16 aPack[16 * 32 * 16];          // [ks][lane][16], 16 KB

  int lane = threadIdx.x & 31;
  int cb   = threadIdx.x >> 5;    // column block 0..31
  int hf   = lane >> 4;
  int n    = lane & 15;
  int jcol = cb * 16 + n;         // h column owned by this lane's C elements

  // A-layout coordinates of column jcol (constant per lane):
  int Kp    = jcol & 31;
  int halfA = (Kp >> 3) & 1;
  int jA    = (Kp & 7) | ((Kp & 16) >> 1);
  int ksA   = jcol >> 5;

  v8f h = {};
  if (h0) {
#pragma unroll
    for (int r = 0; r < 8; ++r) h[r] = h0[(r + 8 * hf) * H_ + jcol];
  }
  // prologue: publish h into aPack
#pragma unroll
  for (int r = 0; r < 8; ++r) {
    int laneA = (r + 8 * hf) + 16 * halfA;
    aPack[(ksA * 32 + laneA) * 16 + jA] = (__bf16)h[r];
  }

  const __bf16* wr = Whh + (long long)cb * 8192;
  const __bf16* wz = Whh + (long long)(cb + 32) * 8192;
  const __bf16* wn = Whh + (long long)(cb + 64) * 8192;
  float br = bhh[jcol], bz = bhh[H_ + jcol], bn = bhh[2 * H_ + jcol];

  for (int t = 0; t < steps; ++t) {
    __syncthreads();              // aPack writes visible
    v8f ar = {}, az = {}, an = {};
#pragma unroll
    for (int ksI = 0; ksI < 16; ++ksI) {
      bfx16 a  = *(const bfx16*)(aPack + (ksI * 32 + lane) * 16);
      bfx16 b0 = *(const bfx16*)(wr + (ksI * 32 + lane) * 16);
      bfx16 b1 = *(const bfx16*)(wz + (ksI * 32 + lane) * 16);
      bfx16 b2 = *(const bfx16*)(wn + (ksI * 32 + lane) * 16);
      ar = wmma_bf16(a, b0, ar);
      az = wmma_bf16(a, b1, az);
      an = wmma_bf16(a, b2, an);
    }
    const float* gxt = gx + (long long)t * (B_ * G3H);
#pragma unroll
    for (int r = 0; r < 8; ++r) {
      int b = r + 8 * hf;
      const float* gb = gxt + b * G3H;
      float rg = sig_(gb[jcol] + ar[r] + br);
      float ug = sig_(gb[H_ + jcol] + az[r] + bz);
      float ng = tanhf(gb[2 * H_ + jcol] + rg * (an[r] + bn));
      h[r] = (1.0f - ug) * ng + ug * h[r];
    }
    __syncthreads();              // all waves done reading aPack
#pragma unroll
    for (int r = 0; r < 8; ++r) {
      int laneA = (r + 8 * hf) + 16 * halfA;
      int idx = (ksA * 32 + laneA) * 16 + jA;
      __bf16 v = (__bf16)h[r];
      aPack[idx] = v;
      if (hsA) hsA[(long long)t * 8192 + idx] = v;   // hs[t] pre-packed for logits GEMM
    }
  }
  if (hOut) {
#pragma unroll
    for (int r = 0; r < 8; ++r) hOut[(r + 8 * hf) * H_ + jcol] = h[r];
  }
}

// ---------------------------------------------------------------------------
// z = h_last @ fc_enc_W^T + b   (tiny: 2048 dots of 512)
// ---------------------------------------------------------------------------
__global__ void k_z(const float* __restrict__ h, const float* __restrict__ W,
                    const float* __restrict__ b, float* __restrict__ z) {
  int tid = blockIdx.x * blockDim.x + threadIdx.x;
  if (tid >= B_ * L_) return;
  int bi = tid >> 7, o = tid & 127;
  const float* hr = h + bi * H_;
  const float* wr = W + o * H_;
  float s = b[o];
  for (int k = 0; k < H_; ++k) s += hr[k] * wr[k];
  z[tid] = s;
}

// hid0 = tanh(z @ fc_dec_W^T + b)
__global__ void k_hid0(const float* __restrict__ z, const float* __restrict__ W,
                       const float* __restrict__ b, float* __restrict__ h0) {
  int tid = blockIdx.x * blockDim.x + threadIdx.x;
  if (tid >= B_ * H_) return;
  int bi = tid >> 9, o = tid & 511;
  const float* zr = z + bi * L_;
  const float* wr = W + o * L_;
  float s = b[o];
  for (int k = 0; k < L_; ++k) s += zr[k] * wr[k];
  h0[tid] = tanhf(s);
}

// outputs[:,0,:] = 0
__global__ void k_zero_t0(float* __restrict__ out) {
  long long tid = (long long)blockIdx.x * blockDim.x + threadIdx.x;
  if (tid >= (long long)B_ * V_) return;
  long long b = tid / V_, v = tid % V_;
  out[b * (long long)T_ * V_ + v] = 0.0f;
}

// ---------------------------------------------------------------------------
// logits[b, t+1, v] = hs[t,b,:] . dec_fc_W[v,:] + bias[v]
// One wave per 16x16 tile; M-tile = timestep t (its rows are the 16 batches);
// K=512 -> 16 bf16 WMMAs. Both operands pre-packed -> one 32B load each.
// ---------------------------------------------------------------------------
__global__ __launch_bounds__(128)
void k_logits(const __bf16* __restrict__ hsA,   // [127][8192]
              const __bf16* __restrict__ Wp,    // [2000][8192]
              const float* __restrict__ bias,   // [V]
              float* __restrict__ out) {        // [B,T,V]
  int lane = threadIdx.x & 31;
  int nt = blockIdx.x * 4 + (threadIdx.x >> 5); // vocab tile 0..1999
  int mt = blockIdx.y;                          // timestep 0..126
  const __bf16* aB = hsA + (long long)mt * 8192;
  const __bf16* bB = Wp + (long long)nt * 8192;
  v8f acc = {};
#pragma unroll
  for (int ksI = 0; ksI < 16; ++ksI) {
    bfx16 a = *(const bfx16*)(aB + (ksI * 32 + lane) * 16);
    bfx16 b = *(const bfx16*)(bB + (ksI * 32 + lane) * 16);
    if (ksI + 1 < 16)
      __builtin_prefetch(bB + ((ksI + 1) * 32 + lane) * 16, 0, 1);
    acc = wmma_bf16(a, b, acc);
  }
  int v = nt * 16 + (lane & 15);
  float bb = bias[v];
  int hf = lane >> 4;
#pragma unroll
  for (int r = 0; r < 8; ++r) {
    int bI = r + 8 * hf;                         // batch
    out[((long long)bI * T_ + (mt + 1)) * V_ + v] = acc[r] + bb;
  }
}

// ---------------------------------------------------------------------------
extern "C" void kernel_launch(void* const* d_in, const int* in_sizes, int n_in,
                              void* d_out, int out_size, void* d_ws, size_t ws_size,
                              hipStream_t stream) {
  (void)in_sizes; (void)n_in; (void)out_size; (void)ws_size;
  const int*   x        = (const int*)d_in[0];
  const float* enc_emb  = (const float*)d_in[1];
  const float* enc_W_ih = (const float*)d_in[2];
  const float* enc_b_ih = (const float*)d_in[3];
  const float* enc_W_hh = (const float*)d_in[4];
  const float* enc_b_hh = (const float*)d_in[5];
  const float* fc_enc_W = (const float*)d_in[6];
  const float* fc_enc_b = (const float*)d_in[7];
  const float* fc_dec_W = (const float*)d_in[8];
  const float* fc_dec_b = (const float*)d_in[9];
  const float* dec_emb  = (const float*)d_in[10];
  const float* dec_W_ih = (const float*)d_in[11];
  const float* dec_b_ih = (const float*)d_in[12];
  const float* dec_W_hh = (const float*)d_in[13];
  const float* dec_b_hh = (const float*)d_in[14];
  const float* dec_fc_W = (const float*)d_in[15];
  const float* dec_fc_b = (const float*)d_in[16];
  float* out = (float*)d_out;

  // workspace carve-out (~65 MB total)
  char* ws = (char*)d_ws;
  auto carve = [&](size_t bytes) -> char* {
    char* p = ws; ws += (bytes + 255) & ~(size_t)255; return p;
  };
  __bf16* WpEncIh = (__bf16*)carve((size_t)G3H * E_ * 2);
  __bf16* WpEncHh = (__bf16*)carve((size_t)G3H * H_ * 2);
  __bf16* WpDecIh = (__bf16*)carve((size_t)G3H * E_ * 2);
  __bf16* WpDecHh = (__bf16*)carve((size_t)G3H * H_ * 2);
  __bf16* WpFc    = (__bf16*)carve((size_t)V_ * H_ * 2);
  float*  gxEnc   = (float*)carve((size_t)T_ * B_ * G3H * 4);
  float*  gxDec   = (float*)carve((size_t)(T_ - 1) * B_ * G3H * 4);
  float*  hEnc    = (float*)carve((size_t)B_ * H_ * 4);
  float*  h0Dec   = (float*)carve((size_t)B_ * H_ * 4);
  __bf16* hsA     = (__bf16*)carve((size_t)(T_ - 1) * 8192 * 2);
  float*  zOut    = out + (size_t)B_ * T_ * V_;   // second output: z [B,L]

  auto pack = [&](const float* W, __bf16* o, int N, int K) {
    long long tot = (long long)N * K;
    k_pack_b<<<dim3((unsigned)((tot + 255) / 256)), dim3(256), 0, stream>>>(W, o, N, K);
  };
  pack(enc_W_ih, WpEncIh, G3H, E_);
  pack(enc_W_hh, WpEncHh, G3H, H_);
  pack(dec_W_ih, WpDecIh, G3H, E_);
  pack(dec_W_hh, WpDecHh, G3H, H_);
  pack(dec_fc_W, WpFc,    V_,  H_);

  // encoder input projection + scan
  k_gx<<<dim3(NT3H, T_), dim3(32), 0, stream>>>(x, enc_emb, WpEncIh, enc_b_ih, gxEnc);
  k_gru_scan<<<dim3(1), dim3(1024), 0, stream>>>(gxEnc, WpEncHh, enc_b_hh,
                                                 nullptr, T_, nullptr, hEnc);
  // bottleneck
  k_z<<<dim3((B_ * L_ + 255) / 256), dim3(256), 0, stream>>>(hEnc, fc_enc_W, fc_enc_b, zOut);
  k_hid0<<<dim3((B_ * H_ + 255) / 256), dim3(256), 0, stream>>>(zOut, fc_dec_W, fc_dec_b, h0Dec);

  // decoder input projection + scan (teacher forcing: tokens x[:, :T-1])
  k_gx<<<dim3(NT3H, T_ - 1), dim3(32), 0, stream>>>(x, dec_emb, WpDecIh, dec_b_ih, gxDec);
  k_gru_scan<<<dim3(1), dim3(1024), 0, stream>>>(gxDec, WpDecHh, dec_b_hh,
                                                 h0Dec, T_ - 1, hsA, nullptr);

  // outputs
  k_zero_t0<<<dim3((B_ * V_ + 255) / 256), dim3(256), 0, stream>>>(out);
  k_logits<<<dim3(NTV / 4, T_ - 1), dim3(128), 0, stream>>>(hsA, WpFc, dec_fc_b, out);
}
